// SAGE_87540023427925
// MI455X (gfx1250) — matrix operand order repaired and checked by reading, not compiled, not run on hardware
//
#include <hip/hip_runtime.h>

typedef __attribute__((ext_vector_type(16))) _Float16 v16h;
typedef __attribute__((ext_vector_type(8)))  float    v8f;

#define F 128  // feature dim (fixed: IN=HID=OUT=128)

// ---------------------------------------------------------------- zero fill
__global__ void zero_kernel(float* __restrict__ p, int n4) {
    int i = blockIdx.x * blockDim.x + threadIdx.x;
    if (i < n4) ((float4*)p)[i] = make_float4(0.f, 0.f, 0.f, 0.f);
}

// ------------------------------------------------- A fragment (16x32 f16)
// Layout used here (contiguous k, same as B): lane m=lane&15 holds row M=m; kh=lane>>4.
// elements 0..15 -> K = k0 + kh*16 + e
__device__ inline v16h load_a_frag(const float* __restrict__ arow, int k0, int kh) {
    const float4* p0 = (const float4*)(arow + k0 + kh * 16);
    const float4* p1 = (const float4*)(arow + k0 + kh * 16 + 8);
    float4 f0 = p0[0], f1 = p0[1], f2 = p1[0], f3 = p1[1];
    v16h a;
    a[0] = (_Float16)f0.x; a[1] = (_Float16)f0.y; a[2]  = (_Float16)f0.z; a[3]  = (_Float16)f0.w;
    a[4] = (_Float16)f1.x; a[5] = (_Float16)f1.y; a[6]  = (_Float16)f1.z; a[7]  = (_Float16)f1.w;
    a[8] = (_Float16)f2.x; a[9] = (_Float16)f2.y; a[10] = (_Float16)f2.z; a[11] = (_Float16)f2.w;
    a[12]= (_Float16)f3.x; a[13]= (_Float16)f3.y; a[14] = (_Float16)f3.z; a[15] = (_Float16)f3.w;
    return a;
}

// C = [relu]( A1 @ W1 (+ A2 @ W2) + bias ),  A:[N,128] f32, W:[128,128] f32.
// Block = 256 threads = 8 waves; wave w owns 16 output columns c0=w*16.
// Blocks grid-stride over 16-row tiles; weight fragments stay in registers.
__global__ __launch_bounds__(256) void gemm_kernel(
    const float* __restrict__ A1, const float* __restrict__ W1,
    const float* __restrict__ A2, const float* __restrict__ W2,
    const float* __restrict__ bias, float* __restrict__ C,
    int nRowTiles, int do_relu)
{
    const int lane = threadIdx.x & 31;
    const int wv   = threadIdx.x >> 5;
    const int m    = lane & 15;
    const int kh   = lane >> 4;
    const int c0   = wv * 16;

    // Preload B fragments (32x16 f16): lane col N=m, element e -> K = kh*16 + e
    v16h bf1[4], bf2[4];
    #pragma unroll
    for (int kc = 0; kc < 4; ++kc) {
        const float* wp = W1 + (size_t)(kc * 32 + kh * 16) * F + c0 + m;
        #pragma unroll
        for (int e = 0; e < 16; ++e) bf1[kc][e] = (_Float16)wp[e * F];
    }
    if (A2) {
        #pragma unroll
        for (int kc = 0; kc < 4; ++kc) {
            const float* wp = W2 + (size_t)(kc * 32 + kh * 16) * F + c0 + m;
            #pragma unroll
            for (int e = 0; e < 16; ++e) bf2[kc][e] = (_Float16)wp[e * F];
        }
    }
    const float bb = bias[c0 + m];

    for (int t = blockIdx.x; t < nRowTiles; t += gridDim.x) {
        const int r0 = t * 16;
        v8f acc = {};
        const float* arow = A1 + (size_t)(r0 + m) * F;
        #pragma unroll
        for (int kc = 0; kc < 4; ++kc) {
            v16h a = load_a_frag(arow, kc * 32, kh);
            acc = __builtin_amdgcn_wmma_f32_16x16x32_f16(
                false, a, false, bf1[kc], (short)0, acc, false, false);
        }
        if (A2) {
            const float* arow2 = A2 + (size_t)(r0 + m) * F;
            #pragma unroll
            for (int kc = 0; kc < 4; ++kc) {
                v16h a = load_a_frag(arow2, kc * 32, kh);
                acc = __builtin_amdgcn_wmma_f32_16x16x32_f16(
                    false, a, false, bf2[kc], (short)0, acc, false, false);
            }
        }
        // C/D layout: element r -> row M = r + 8*kh, col N = m
        #pragma unroll
        for (int r = 0; r < 8; ++r) {
            float v = acc[r] + bb;
            if (do_relu) v = fmaxf(v, 0.f);
            C[(size_t)(r0 + r + kh * 8) * F + c0 + m] = v;
        }
    }
}

// ------------------------------------------- edge segment-max (one edge/wave)
// y >= 0 (post-relu) so float max == unsigned-int max on raw bits.
__global__ __launch_bounds__(256) void edge_max_kernel(
    const float* __restrict__ y, const int* __restrict__ src,
    const int* __restrict__ dst, float* __restrict__ agg, int E)
{
    int e = blockIdx.x * 8 + (threadIdx.x >> 5);
    if (e >= E) return;
    int lane = threadIdx.x & 31;
    int s = src[e], d = dst[e];
    float4 v = ((const float4*)(y + (size_t)s * F))[lane];
    unsigned* out = (unsigned*)(agg + (size_t)d * F) + lane * 4;
    atomicMax(out + 0, __float_as_uint(v.x));
    atomicMax(out + 1, __float_as_uint(v.y));
    atomicMax(out + 2, __float_as_uint(v.z));
    atomicMax(out + 3, __float_as_uint(v.w));
}

// -------------------------------------- row L2 normalize + relu (row/wave)
__global__ __launch_bounds__(256) void norm_relu_kernel(
    const float* __restrict__ in, float* __restrict__ out, int N)
{
    int row = blockIdx.x * 8 + (threadIdx.x >> 5);
    if (row >= N) return;
    int lane = threadIdx.x & 31;
    float4 v = ((const float4*)(in + (size_t)row * F))[lane];
    float s = v.x * v.x + v.y * v.y + v.z * v.z + v.w * v.w;
    #pragma unroll
    for (int off = 16; off > 0; off >>= 1) s += __shfl_xor(s, off, 32);
    float inv = 1.0f / fmaxf(sqrtf(s), 1e-12f);  // relu(h/n) == max(h,0)/n, n>0
    v.x = fmaxf(v.x, 0.f) * inv;
    v.y = fmaxf(v.y, 0.f) * inv;
    v.z = fmaxf(v.z, 0.f) * inv;
    v.w = fmaxf(v.w, 0.f) * inv;
    ((float4*)(out + (size_t)row * F))[lane] = v;
}

extern "C" void kernel_launch(void* const* d_in, const int* in_sizes, int n_in,
                              void* d_out, int out_size, void* d_ws, size_t ws_size,
                              hipStream_t stream) {
    const float* x   = (const float*)d_in[0];
    const int*   src = (const int*)  d_in[1];
    const int*   dst = (const int*)  d_in[2];
    const float* Wp1 = (const float*)d_in[3];
    const float* bp1 = (const float*)d_in[4];
    const float* Ws1 = (const float*)d_in[5];
    const float* Wn1 = (const float*)d_in[6];
    const float* b1  = (const float*)d_in[7];
    const float* Wp2 = (const float*)d_in[8];
    const float* bp2 = (const float*)d_in[9];
    const float* Ws2 = (const float*)d_in[10];
    const float* Wn2 = (const float*)d_in[11];
    const float* b2  = (const float*)d_in[12];

    const int N = in_sizes[0] / F;   // 50000
    const int E = in_sizes[1];       // 1.6M

    float* outp = (float*)d_out;
    float* y    = (float*)d_ws;                 // [N,128] pooled features
    float* agg  = y + (size_t)N * F;            // [N,128] segment max

    const int rowTiles = (N + 15) / 16;         // 3125 (exact)
    const int gemmGrid = 512;                   // persistent blocks, weights in regs
    const int zeroGrid = ((N * F / 4) + 255) / 256;
    const int edgeGrid = (E + 7) / 8;
    const int rowGrid  = (N + 7) / 8;

    // ---------------- layer 1 ----------------
    zero_kernel<<<zeroGrid, 256, 0, stream>>>(agg, N * F / 4);
    gemm_kernel<<<gemmGrid, 256, 0, stream>>>(x, Wp1, nullptr, nullptr, bp1, y, rowTiles, 1);
    edge_max_kernel<<<edgeGrid, 256, 0, stream>>>(y, src, dst, agg, E);
    gemm_kernel<<<gemmGrid, 256, 0, stream>>>(x, Ws1, agg, Wn1, b1, outp, rowTiles, 0);
    norm_relu_kernel<<<rowGrid, 256, 0, stream>>>(outp, outp, N);   // h1 lives in d_out

    // ---------------- layer 2 ----------------
    zero_kernel<<<zeroGrid, 256, 0, stream>>>(agg, N * F / 4);
    gemm_kernel<<<gemmGrid, 256, 0, stream>>>(outp, Wp2, nullptr, nullptr, bp2, y, rowTiles, 1);
    edge_max_kernel<<<edgeGrid, 256, 0, stream>>>(y, src, dst, agg, E);
    gemm_kernel<<<gemmGrid, 256, 0, stream>>>(outp, Ws2, agg, Wn2, b2, y, rowTiles, 0);
    norm_relu_kernel<<<rowGrid, 256, 0, stream>>>(y, outp, N);
}
